// CompontStyleFusion_71846212927823
// MI455X (gfx1250) — matrix operand order repaired and checked by reading, not compiled, not run on hardware
//
#include <hip/hip_runtime.h>
#include <hip/hip_bf16.h>

typedef __attribute__((ext_vector_type(16))) _Float16 v16h;
typedef __attribute__((ext_vector_type(8)))  float    v8f;

#define IMGS   256   // B*NREF
#define HWN    256   // 16x16 spatial
#define EDIMC  256
#define COUTC  256
#define BATCH  64
#define LPAD   48    // 45 padded to 3 WMMA tiles
#define NKEY   1024  // NREF*HW

// ---------------------------------------------------------------------------
// CDNA5 data-mover feature detection (compile-safe fallback chain)
// ---------------------------------------------------------------------------
#if defined(__has_builtin)
#if __has_builtin(__builtin_amdgcn_tensor_load_to_lds) && __has_builtin(__builtin_amdgcn_s_wait_tensorcnt)
#define HAVE_TDM 1
#endif
#if __has_builtin(__builtin_amdgcn_global_load_async_to_lds_b128) && __has_builtin(__builtin_amdgcn_s_wait_asynccnt)
#define HAVE_ASYNC 1
#endif
#endif

#ifdef HAVE_TDM
typedef __attribute__((ext_vector_type(4))) unsigned int u32x4;
typedef __attribute__((ext_vector_type(8))) int          i32x8;
typedef __attribute__((ext_vector_type(4))) int          i32x4;

// Issue one TDM descriptor moving a contiguous 16KB block (4096 x 4B) from
// global memory to LDS. D# layout per CDNA5 ISA ch.8:
//  g0: [1:0]=count=1 | lds_addr | global_addr[31:0] | global_addr[56:32] + type=2
//  g1: data_size=2(4B); tensor_dim0=tile_dim0=4096; tensor_dim1=1; stride0=4096
__device__ __forceinline__ void tdm_load_16kb(const void* gsrc, unsigned lds_off) {
    unsigned long long ga = (unsigned long long)gsrc;
    u32x4 g0 = { 1u,                                   // count=1, user mode
                 lds_off,                              // lds_addr (bytes)
                 (unsigned)ga,                         // global_addr[31:0]
                 ((unsigned)(ga >> 32) & 0x01FFFFFFu) | (2u << 30) }; // addr[56:32]|type=2
    i32x8 g1 = { 0x00020000,   // wg_mask=0, data_size=2 (4 bytes)
                 0x10000000,   // tensor_dim0 = 4096 (bits 79:48, low 16 here)
                 0x00010000,   // tensor_dim1 = 1
                 0x10000000,   // tile_dim0 = 4096 (bits 127:112)
                 0,            // tile_dim1/2 unused
                 4096, 0, 0 }; // tensor_dim0_stride = 4096
    i32x4 z4 = { 0, 0, 0, 0 };
#if __clang_major__ >= 23
    i32x8 z8 = { 0, 0, 0, 0, 0, 0, 0, 0 };
    __builtin_amdgcn_tensor_load_to_lds(g0, g1, z4, z4, z8, 0);
#else
    __builtin_amdgcn_tensor_load_to_lds(g0, g1, z4, z4, 0);
#endif
}
#endif

// ---------------------------------------------------------------------------
// WMMA fragment helpers (gfx1250 16x16x32 f16, wave32)
// A (16x32, MxK): lane L row M=L&15; e -> K=(L>>4)*16+e
// B (32x16, KxN): lane L col N=L&15; e -> K=(L>>4)*16+e
// C/D (16x16 f32): lane L col N=L&15, vgpr r -> row M=r+8*(L>>4)
// ---------------------------------------------------------------------------
__device__ __forceinline__ v8f wmma_f16(v16h a, v16h b, v8f c) {
    return __builtin_amdgcn_wmma_f32_16x16x32_f16(false, a, false, b, (short)0, c, false, false);
}

// ---------------------------------------------------------------------------
// 1) codebook gather -> emb[img][c][p] (f16)
// ---------------------------------------------------------------------------
__global__ __launch_bounds__(256) void gather_embed(const int* __restrict__ idx,
                                                    const float* __restrict__ cb,
                                                    _Float16* __restrict__ emb) {
    int img = blockIdx.x;
    int p   = threadIdx.x;
    int id  = idx[img * HWN + p];
    const float* row = cb + (size_t)id * EDIMC;
    _Float16* op = emb + (size_t)img * EDIMC * HWN + p;
    for (int c = 0; c < EDIMC; ++c) op[(size_t)c * HWN] = (_Float16)row[c];
}

// ---------------------------------------------------------------------------
// 2) 3x3 conv (reflect pad 1) + instance-norm (+optional skip) + relu, WMMA.
//    One workgroup = one image x 16 output channels x full 256 spatial.
//    Input staged to LDS in 32-channel chunks (16KB), TDM double-buffered.
// ---------------------------------------------------------------------------
__device__ __forceinline__ void conv_chunk_mma(const _Float16* __restrict__ sIn,
                                               v8f (&acc)[4],
                                               const float* __restrict__ w,
                                               int Cin, int cout0, int cin0,
                                               int wave, int half, int l16) {
    for (int tap = 0; tap < 9; ++tap) {
        const int dy = tap / 3 - 1, dx = tap % 3 - 1;
        // A fragment: weights [cout0+m][cin0+k][tap], f32 -> f16
        v16h af;
        {
            const float* wp = w + ((size_t)(cout0 + l16) * Cin + cin0) * 9 + tap;
            const int kb = half * 16;
#pragma unroll
            for (int e = 0; e < 8; ++e) af[e] = (_Float16)wp[(size_t)(kb + e) * 9];
#pragma unroll
            for (int e = 0; e < 8; ++e) af[8 + e] = (_Float16)wp[(size_t)(kb + 8 + e) * 9];
        }
#pragma unroll
        for (int t = 0; t < 4; ++t) {
            const int n = (wave * 4 + t) * 16 + l16;   // spatial position
            int h  = (n >> 4) + dy;
            int wx = (n & 15) + dx;
            h  = h  < 0 ? -h  : (h  > 15 ? 30 - h  : h);
            wx = wx < 0 ? -wx : (wx > 15 ? 30 - wx : wx);
            const int sp = h * 16 + wx;
            const int kb = half * 16;
            v16h bf;
#pragma unroll
            for (int e = 0; e < 16; ++e) bf[e] = sIn[(kb + e) * HWN + sp];
            acc[t] = wmma_f16(af, bf, acc[t]);
        }
    }
}

__global__ __launch_bounds__(128) void conv3x3_wmma(const _Float16* __restrict__ in,
                                                    const float* __restrict__ w,
                                                    const _Float16* __restrict__ skip,
                                                    _Float16* __restrict__ out,
                                                    int Cin, int Cout) {
#ifdef HAVE_TDM
    __shared__ _Float16 sIn[2][32 * HWN];  // 2 x 16KB ping-pong staging
#else
    __shared__ _Float16 sIn[1][32 * HWN];  // 16KB staging
#endif
    __shared__ float sTile[16 * HWN];      // 16KB epilogue tile
    __shared__ float sMean[16], sInv[16];

    const int img   = blockIdx.y;
    const int cout0 = blockIdx.x * 16;
    const int tid   = threadIdx.x;
    const int wave  = tid >> 5;
    const int lane  = tid & 31;
    const int half  = lane >> 4;
    const int l16   = lane & 15;

    v8f zero = {};
    v8f acc[4] = {zero, zero, zero, zero};

    const _Float16* inImg = in + (size_t)img * Cin * HWN;
    const int nChunk = Cin >> 5;

#ifdef HAVE_TDM
    // Tensor Data Mover: DMA chunk c+1 into the other LDS buffer while the
    // WMMA pipeline consumes chunk c. Dependencies via TENSORcnt + barriers.
    const unsigned ldsBase = (unsigned)(unsigned long long)(const void*)&sIn[0][0];
    if (wave == 0) tdm_load_16kb(inImg, ldsBase);
    __builtin_amdgcn_s_wait_tensorcnt(0);
    __syncthreads();
    for (int c = 0; c < nChunk; ++c) {
        if (c + 1 < nChunk && wave == 0)
            tdm_load_16kb(inImg + (size_t)(c + 1) * 32 * HWN,
                          ldsBase + (unsigned)(((c + 1) & 1) * (32 * HWN * 2)));
        conv_chunk_mma(sIn[c & 1], acc, w, Cin, cout0, c * 32, wave, half, l16);
        __syncthreads();                        // all waves done reading sIn[c&1]
        __builtin_amdgcn_s_wait_tensorcnt(0);   // chunk c+1 landed (wave 0)
        __syncthreads();                        // LDS visibility for all waves
    }
#else
    for (int c = 0; c < nChunk; ++c) {
        const int cin0 = c * 32;
#ifdef HAVE_ASYNC
        {
            typedef __attribute__((address_space(1))) char gchar;
            typedef __attribute__((address_space(3))) char lchar;
            gchar* gp = (gchar*)(inImg + (size_t)cin0 * HWN);
            lchar* lp = (lchar*)&sIn[0][0];
            for (int i = tid; i < (32 * HWN) / 8; i += 128)
                __builtin_amdgcn_global_load_async_to_lds_b128(gp + i * 16, lp + i * 16, 0, 0);
            __builtin_amdgcn_s_wait_asynccnt(0);
        }
#else
        if (cin0 + 32 < Cin && lane == 0)
            __builtin_prefetch((const void*)(inImg + (size_t)(cin0 + 32) * HWN), 0, 1);
        {
            const uint4* gsrc = (const uint4*)(inImg + (size_t)cin0 * HWN);
            uint4* ldst = (uint4*)&sIn[0][0];
            for (int i = tid; i < (32 * HWN) / 8; i += 128) ldst[i] = gsrc[i];
        }
#endif
        __syncthreads();
        conv_chunk_mma(sIn[0], acc, w, Cin, cout0, cin0, wave, half, l16);
        __syncthreads();
    }
#endif

    // epilogue: spill tile to LDS, instance-norm per channel, +skip, relu
#pragma unroll
    for (int t = 0; t < 4; ++t)
#pragma unroll
        for (int r = 0; r < 8; ++r)
            sTile[(r + 8 * half) * HWN + (wave * 4 + t) * 16 + l16] = acc[t][r];
    __syncthreads();

    {
        const int ch = tid >> 3, sub = tid & 7;
        float s = 0.f, s2 = 0.f;
        for (int p = sub; p < HWN; p += 8) {
            float v = sTile[ch * HWN + p];
            s += v; s2 += v * v;
        }
        for (int m = 4; m; m >>= 1) { s += __shfl_xor(s, m); s2 += __shfl_xor(s2, m); }
        if (sub == 0) {
            float mu  = s * (1.f / HWN);
            float var = s2 * (1.f / HWN) - mu * mu;
            sMean[ch] = mu;
            sInv[ch]  = rsqrtf(var + 1e-5f);
        }
    }
    __syncthreads();

    {
        _Float16* op = out + (size_t)img * Cout * HWN + (size_t)cout0 * HWN;
        const _Float16* sk = skip ? skip + (size_t)img * Cout * HWN + (size_t)cout0 * HWN
                                  : (const _Float16*)nullptr;
        for (int i = tid; i < 16 * HWN; i += 128) {
            const int ch = i >> 8;
            float v = (sTile[i] - sMean[ch]) * sInv[ch];
            if (sk) v += (float)sk[i];
            v = v > 0.f ? v : 0.f;
            op[i] = (_Float16)v;
        }
    }
}

// ---------------------------------------------------------------------------
// 3) 1x1 conv, raw output (skip path of res block 3), WMMA
// ---------------------------------------------------------------------------
__global__ __launch_bounds__(32) void conv1x1_wmma(const _Float16* __restrict__ in,
                                                   const float* __restrict__ w,
                                                   _Float16* __restrict__ out,
                                                   int Cin, int Cout) {
    const int img = blockIdx.z, cout0 = blockIdx.x * 16, n0 = blockIdx.y * 16;
    const int lane = threadIdx.x & 31, half = lane >> 4, l16 = lane & 15;
    v8f acc = {};
    for (int cin0 = 0; cin0 < Cin; cin0 += 32) {
        v16h af, bf;
        const float* wp = w + (size_t)(cout0 + l16) * Cin + cin0;
        const int kb = half * 16;
#pragma unroll
        for (int e = 0; e < 8; ++e) { af[e] = (_Float16)wp[kb + e]; af[8 + e] = (_Float16)wp[kb + 8 + e]; }
        const _Float16* ip = in + (size_t)img * Cin * HWN + (size_t)cin0 * HWN;
        const int bb = half * 16;
#pragma unroll
        for (int e = 0; e < 16; ++e) bf[e] = ip[(size_t)(bb + e) * HWN + n0 + l16];
        acc = wmma_f16(af, bf, acc);
    }
    _Float16* op = out + (size_t)img * Cout * HWN;
#pragma unroll
    for (int r = 0; r < 8; ++r)
        op[(size_t)(cout0 + r + 8 * half) * HWN + n0 + l16] = (_Float16)acc[r];
}

// ---------------------------------------------------------------------------
// 4) softmax(sim*3)-weighted pooling: xg[b][p][c] (f32)
// ---------------------------------------------------------------------------
__global__ __launch_bounds__(256) void xg_kernel(const float* __restrict__ sim,
                                                 const _Float16* __restrict__ act,
                                                 float* __restrict__ xg) {
    const int b = blockIdx.x, p = threadIdx.x;
    float s[4], mx = -1e30f;
    for (int n = 0; n < 4; ++n) { s[n] = sim[b * 4 + n] * 3.f; mx = fmaxf(mx, s[n]); }
    float sum = 0.f;
    for (int n = 0; n < 4; ++n) { s[n] = __expf(s[n] - mx); sum += s[n]; }
    const float inv = 1.f / sum;
    for (int c = 0; c < COUTC; ++c) {
        float a = 0.f;
        for (int n = 0; n < 4; ++n)
            a += s[n] * inv * (float)act[((size_t)(b * 4 + n) * COUTC + c) * HWN + p];
        xg[((size_t)b * HWN + p) * COUTC + c] = a;
    }
}

// feat[b][m][c] = act[(b*4 + m/256)][c][m%256]
__global__ __launch_bounds__(256) void feat_kernel(const _Float16* __restrict__ act,
                                                   _Float16* __restrict__ feat) {
    const int bm = blockIdx.x, c = threadIdx.x;
    const int b = bm >> 10, m = bm & 1023, n = m >> 8, p = m & 255;
    feat[(size_t)bm * COUTC + c] = act[((size_t)(b * 4 + n) * COUTC + c) * HWN + p];
}

// qin[b][l][c] = ids + wpe (rows 45..47 zero-padded), f16
__global__ __launch_bounds__(256) void prepq_kernel(const float* __restrict__ ids,
                                                    const float* __restrict__ wpe,
                                                    _Float16* __restrict__ qin) {
    const int bl = blockIdx.x, c = threadIdx.x;
    const int b = bl / LPAD, l = bl % LPAD;
    float v = (l < 45) ? ids[((size_t)b * 45 + l) * COUTC + c] + wpe[(size_t)l * COUTC + c] : 0.f;
    qin[(size_t)bl * COUTC + c] = (_Float16)v;
}

// ---------------------------------------------------------------------------
// 5) GEMM: out[b] = A[b](MxK,f16) x W^T (W: NxK f32 shared), WMMA
// ---------------------------------------------------------------------------
template <bool OUTF16>
__global__ __launch_bounds__(32) void gemm_aw(const _Float16* __restrict__ A,
                                              const float* __restrict__ W,
                                              void* __restrict__ outv,
                                              int N, int K,
                                              long long Astride, long long Ostride) {
    const int b = blockIdx.z, n0 = blockIdx.x * 16, m0 = blockIdx.y * 16;
    const int lane = threadIdx.x & 31, half = lane >> 4, l16 = lane & 15;
    const _Float16* Ab = A + (size_t)b * Astride;
    v8f acc = {};
    for (int k0 = 0; k0 < K; k0 += 32) {
        v16h af, bf;
        const _Float16* ap = Ab + (size_t)(m0 + l16) * K + k0;
        const int kb = half * 16;
#pragma unroll
        for (int e = 0; e < 8; ++e) { af[e] = ap[kb + e]; af[8 + e] = ap[kb + 8 + e]; }
        const float* wp = W + (size_t)(n0 + l16) * K + k0;
        const int bb = half * 16;
#pragma unroll
        for (int e = 0; e < 16; ++e) bf[e] = (_Float16)wp[bb + e];
        acc = wmma_f16(af, bf, acc);
    }
    if (OUTF16) {
        _Float16* O = (_Float16*)outv + (size_t)b * Ostride;
#pragma unroll
        for (int r = 0; r < 8; ++r) O[(size_t)(m0 + r + 8 * half) * N + n0 + l16] = (_Float16)acc[r];
    } else {
        float* O = (float*)outv + (size_t)b * Ostride;
#pragma unroll
        for (int r = 0; r < 8; ++r) O[(size_t)(m0 + r + 8 * half) * N + n0 + l16] = acc[r];
    }
}

// ---------------------------------------------------------------------------
// 6) GEMM: out[b] = A[b](MxK,f16) x B[b] (f16; BTRANS: B is [N][K], else [K][N])
// ---------------------------------------------------------------------------
template <bool BTRANS, bool OUTF16>
__global__ __launch_bounds__(32) void gemm_ab(const _Float16* __restrict__ A,
                                              const _Float16* __restrict__ B,
                                              void* __restrict__ outv,
                                              int N, int K,
                                              long long Astride, long long Bstride,
                                              long long Ostride) {
    const int b = blockIdx.z, n0 = blockIdx.x * 16, m0 = blockIdx.y * 16;
    const int lane = threadIdx.x & 31, half = lane >> 4, l16 = lane & 15;
    const _Float16* Ab = A + (size_t)b * Astride;
    const _Float16* Bb = B + (size_t)b * Bstride;
    v8f acc = {};
    for (int k0 = 0; k0 < K; k0 += 32) {
        v16h af, bf;
        const _Float16* ap = Ab + (size_t)(m0 + l16) * K + k0;
        const int kb = half * 16;
#pragma unroll
        for (int e = 0; e < 8; ++e) { af[e] = ap[kb + e]; af[8 + e] = ap[kb + 8 + e]; }
        const int bb = half * 16;
#pragma unroll
        for (int e = 0; e < 16; ++e)
            bf[e] = BTRANS ? Bb[(size_t)(n0 + l16) * K + k0 + bb + e]
                           : Bb[(size_t)(k0 + bb + e) * N + n0 + l16];
        acc = wmma_f16(af, bf, acc);
    }
    if (OUTF16) {
        _Float16* O = (_Float16*)outv + (size_t)b * Ostride;
#pragma unroll
        for (int r = 0; r < 8; ++r) O[(size_t)(m0 + r + 8 * half) * N + n0 + l16] = (_Float16)acc[r];
    } else {
        float* O = (float*)outv + (size_t)b * Ostride;
#pragma unroll
        for (int r = 0; r < 8; ++r) O[(size_t)(m0 + r + 8 * half) * N + n0 + l16] = acc[r];
    }
}

// ---------------------------------------------------------------------------
// 7) per-head layernorms (HD=32 -> one wave per head, shuffle reductions)
// ---------------------------------------------------------------------------
__global__ __launch_bounds__(256) void ln_q_kernel(const float* __restrict__ Q,
                                                   const float* __restrict__ g,
                                                   const float* __restrict__ bia,
                                                   _Float16* __restrict__ qf) {
    const int b = blockIdx.x, l = blockIdx.y;
    const int h = threadIdx.x >> 5, d = threadIdx.x & 31;
    float v = (l < 45) ? Q[((size_t)b * LPAD + l) * COUTC + h * 32 + d] : 0.f;
    float s = v, s2 = v * v;
    for (int m = 16; m; m >>= 1) { s += __shfl_xor(s, m); s2 += __shfl_xor(s2, m); }
    const float mu = s * (1.f / 32.f), var = s2 * (1.f / 32.f) - mu * mu;
    float y = (v - mu) * rsqrtf(var + 1e-5f) * g[d] + bia[d];
    qf[((size_t)(b * 8 + h) * LPAD + l) * 32 + d] = (l < 45) ? (_Float16)y : (_Float16)0.f;
}

__global__ __launch_bounds__(256) void ln_kv_kernel(const _Float16* __restrict__ KV,
                                                    const float* __restrict__ g,
                                                    const float* __restrict__ bia,
                                                    _Float16* __restrict__ kf,
                                                    _Float16* __restrict__ vf) {
    const int b = blockIdx.x, m = blockIdx.y;
    const int h = threadIdx.x >> 5, d = threadIdx.x & 31;
    const size_t base = ((size_t)b * NKEY + m) * (2 * COUTC);
    float kv = (float)KV[base + h * 32 + d];
    float vv = (float)KV[base + COUTC + h * 32 + d];
    float s = kv, s2 = kv * kv;
    for (int msk = 16; msk; msk >>= 1) { s += __shfl_xor(s, msk); s2 += __shfl_xor(s2, msk); }
    const float mu = s * (1.f / 32.f), var = s2 * (1.f / 32.f) - mu * mu;
    float y = (kv - mu) * rsqrtf(var + 1e-5f) * g[d] + bia[d];
    const size_t o = ((size_t)(b * 8 + h) * NKEY + m) * 32 + d;
    kf[o] = (_Float16)y;
    vf[o] = (_Float16)vv;
}

// ---------------------------------------------------------------------------
// 8) row softmax on scores (scaled), rows >=45 -> zeros
// ---------------------------------------------------------------------------
__global__ __launch_bounds__(256) void softmax_rows(const float* __restrict__ S,
                                                    _Float16* __restrict__ P) {
    const int bh = blockIdx.x, r = blockIdx.y, tid = threadIdx.x;
    const float scale = 0.17677669529663687f; // 1/sqrt(32)
    const size_t base = ((size_t)bh * LPAD + r) * NKEY;
    if (r >= 45) {
        for (int i = tid; i < NKEY; i += 256) P[base + i] = (_Float16)0.f;
        return;
    }
    __shared__ float red[256];
    float mx = -1e30f;
    for (int i = tid; i < NKEY; i += 256) mx = fmaxf(mx, S[base + i] * scale);
    red[tid] = mx; __syncthreads();
    for (int s = 128; s; s >>= 1) { if (tid < s) red[tid] = fmaxf(red[tid], red[tid + s]); __syncthreads(); }
    mx = red[0]; __syncthreads();
    float e[4], sum = 0.f;
#pragma unroll
    for (int j = 0; j < 4; ++j) { e[j] = __expf(S[base + tid + j * 256] * scale - mx); sum += e[j]; }
    red[tid] = sum; __syncthreads();
    for (int s = 128; s; s >>= 1) { if (tid < s) red[tid] += red[tid + s]; __syncthreads(); }
    const float inv = 1.f / red[0];
#pragma unroll
    for (int j = 0; j < 4; ++j) P[base + tid + j * 256] = (_Float16)(e[j] * inv);
}

__global__ __launch_bounds__(256) void merge_heads(const _Float16* __restrict__ o,
                                                   _Float16* __restrict__ om) {
    const int b = blockIdx.x, l = blockIdx.y, c = threadIdx.x;
    const int h = c >> 5, d = c & 31;
    om[((size_t)b * LPAD + l) * COUTC + c] = o[((size_t)(b * 8 + h) * LPAD + l) * 32 + d];
}

// ---------------------------------------------------------------------------
// 9) final concat + layernorm (eps 1e-6) -> d_out
// ---------------------------------------------------------------------------
__global__ __launch_bounds__(256) void final_ln(const float* __restrict__ xl,
                                                const float* __restrict__ xg,
                                                const float* __restrict__ g,
                                                float* __restrict__ out) {
    const int b = blockIdx.x, r = blockIdx.y, c = threadIdx.x;
    float v = (r < 45) ? xl[((size_t)b * LPAD + r) * COUTC + c]
                       : xg[((size_t)b * HWN + (r - 45)) * COUTC + c];
    __shared__ float red[256], red2[256];
    red[c] = v; red2[c] = v * v; __syncthreads();
    for (int s = 128; s; s >>= 1) {
        if (c < s) { red[c] += red[c + s]; red2[c] += red2[c + s]; }
        __syncthreads();
    }
    const float mu = red[0] * (1.f / 256.f), var = red2[0] * (1.f / 256.f) - mu * mu;
    out[((size_t)b * 301 + r) * COUTC + c] = (v - mu) * rsqrtf(var + 1e-6f) * g[c];
}

// ---------------------------------------------------------------------------
extern "C" void kernel_launch(void* const* d_in, const int* in_sizes, int n_in,
                              void* d_out, int out_size, void* d_ws, size_t ws_size,
                              hipStream_t stream) {
    const int*   indices = (const int*)d_in[0];
    const float* ids     = (const float*)d_in[1];
    const float* sim     = (const float*)d_in[2];
    const float* cb      = (const float*)d_in[3];
    const float* w_c1    = (const float*)d_in[4];
    const float* w_c2    = (const float*)d_in[5];
    const float* w_c3    = (const float*)d_in[6];
    const float* r1_w1   = (const float*)d_in[7];
    const float* r1_w2   = (const float*)d_in[8];
    const float* r2_w1   = (const float*)d_in[9];
    const float* r2_w2   = (const float*)d_in[10];
    const float* r3_w1   = (const float*)d_in[11];
    const float* r3_w2   = (const float*)d_in[12];
    const float* r3_skip = (const float*)d_in[13];
    const float* r4_w1   = (const float*)d_in[14];
    const float* r4_w2   = (const float*)d_in[15];
    const float* wpe     = (const float*)d_in[16];
    const float* w_q     = (const float*)d_in[17];
    const float* w_kv    = (const float*)d_in[18];
    const float* w_proj  = (const float*)d_in[19];
    const float* g_lnq   = (const float*)d_in[20];
    const float* b_lnq   = (const float*)d_in[21];
    const float* g_lnk   = (const float*)d_in[22];
    const float* b_lnk   = (const float*)d_in[23];
    const float* g_ln    = (const float*)d_in[24];
    float* out = (float*)d_out;

    char* wsp = (char*)d_ws;
    size_t off = 0;
    auto alloc = [&](size_t bytes) -> void* {
        void* p = wsp + off;
        off += (bytes + 255) & ~(size_t)255;
        return p;
    };

    _Float16* emb  = (_Float16*)alloc((size_t)IMGS * EDIMC * HWN * 2);
    _Float16* a1   = (_Float16*)alloc((size_t)IMGS * 32  * HWN * 2);
    _Float16* a2   = (_Float16*)alloc((size_t)IMGS * 64  * HWN * 2);
    _Float16* a3   = (_Float16*)alloc((size_t)IMGS * 128 * HWN * 2);
    _Float16* hbuf = (_Float16*)alloc((size_t)IMGS * 256 * HWN * 2);
    _Float16* o1   = (_Float16*)alloc((size_t)IMGS * 128 * HWN * 2);
    _Float16* o2   = (_Float16*)alloc((size_t)IMGS * 128 * HWN * 2);
    _Float16* s3   = (_Float16*)alloc((size_t)IMGS * 256 * HWN * 2);
    _Float16* o3   = (_Float16*)alloc((size_t)IMGS * 256 * HWN * 2);
    _Float16* act4 = (_Float16*)alloc((size_t)IMGS * 256 * HWN * 2);
    float*    xg   = (float*)   alloc((size_t)BATCH * HWN * COUTC * 4);
    _Float16* feat = (_Float16*)alloc((size_t)BATCH * NKEY * COUTC * 2);
    _Float16* qin  = (_Float16*)alloc((size_t)BATCH * LPAD * COUTC * 2);
    float*    qf32 = (float*)   alloc((size_t)BATCH * LPAD * COUTC * 4);
    _Float16* kv16 = (_Float16*)alloc((size_t)BATCH * NKEY * 2 * COUTC * 2);
    _Float16* qf16 = (_Float16*)alloc((size_t)BATCH * 8 * LPAD * 32 * 2);
    _Float16* kf16 = (_Float16*)alloc((size_t)BATCH * 8 * NKEY * 32 * 2);
    _Float16* vf16 = (_Float16*)alloc((size_t)BATCH * 8 * NKEY * 32 * 2);
    float*    scor = (float*)   alloc((size_t)BATCH * 8 * LPAD * NKEY * 4);
    _Float16* pmat = (_Float16*)alloc((size_t)BATCH * 8 * LPAD * NKEY * 2);
    _Float16* ofr  = (_Float16*)alloc((size_t)BATCH * 8 * LPAD * 32 * 2);
    _Float16* om   = (_Float16*)alloc((size_t)BATCH * LPAD * COUTC * 2);
    float*    xl   = (float*)   alloc((size_t)BATCH * LPAD * COUTC * 4);

    // ---- conv stack (all L2-resident f16 activations, WMMA implicit GEMM) ----
    gather_embed<<<IMGS, 256, 0, stream>>>(indices, cb, emb);
    conv3x3_wmma<<<dim3(2, IMGS),  128, 0, stream>>>(emb,  w_c1, nullptr, a1,   256, 32);
    conv3x3_wmma<<<dim3(4, IMGS),  128, 0, stream>>>(a1,   w_c2, nullptr, a2,   32,  64);
    conv3x3_wmma<<<dim3(8, IMGS),  128, 0, stream>>>(a2,   w_c3, nullptr, a3,   64,  128);
    conv3x3_wmma<<<dim3(8, IMGS),  128, 0, stream>>>(a3,   r1_w1, nullptr, hbuf, 128, 128);
    conv3x3_wmma<<<dim3(8, IMGS),  128, 0, stream>>>(hbuf, r1_w2, a3,      o1,   128, 128);
    conv3x3_wmma<<<dim3(8, IMGS),  128, 0, stream>>>(o1,   r2_w1, nullptr, hbuf, 128, 128);
    conv3x3_wmma<<<dim3(8, IMGS),  128, 0, stream>>>(hbuf, r2_w2, o1,      o2,   128, 128);
    conv3x3_wmma<<<dim3(16, IMGS), 128, 0, stream>>>(o2,   r3_w1, nullptr, hbuf, 128, 256);
    conv1x1_wmma<<<dim3(16, 16, IMGS), 32, 0, stream>>>(o2, r3_skip, s3, 128, 256);
    conv3x3_wmma<<<dim3(16, IMGS), 128, 0, stream>>>(hbuf, r3_w2, s3,      o3,   256, 256);
    conv3x3_wmma<<<dim3(16, IMGS), 128, 0, stream>>>(o3,   r4_w1, nullptr, hbuf, 256, 256);
    conv3x3_wmma<<<dim3(16, IMGS), 128, 0, stream>>>(hbuf, r4_w2, o3,      act4, 256, 256);

    // ---- pooling + attention ----
    xg_kernel<<<BATCH, 256, 0, stream>>>(sim, act4, xg);
    feat_kernel<<<BATCH * NKEY, 256, 0, stream>>>(act4, feat);
    prepq_kernel<<<BATCH * LPAD, 256, 0, stream>>>(ids, wpe, qin);

    gemm_aw<false><<<dim3(16, 3, BATCH), 32, 0, stream>>>(qin, w_q, qf32,
        COUTC, COUTC, (long long)LPAD * COUTC, (long long)LPAD * COUTC);
    gemm_aw<true><<<dim3(32, 64, BATCH), 32, 0, stream>>>(feat, w_kv, kv16,
        2 * COUTC, COUTC, (long long)NKEY * COUTC, (long long)NKEY * 2 * COUTC);

    ln_q_kernel<<<dim3(BATCH, LPAD), 256, 0, stream>>>(qf32, g_lnq, b_lnq, qf16);
    ln_kv_kernel<<<dim3(BATCH, NKEY), 256, 0, stream>>>(kv16, g_lnk, b_lnk, kf16, vf16);

    // scores[bh] = Q(48x32) x K^T(32x1024)
    gemm_ab<true, false><<<dim3(64, 3, BATCH * 8), 32, 0, stream>>>(qf16, kf16, scor,
        NKEY, 32, (long long)LPAD * 32, (long long)NKEY * 32, (long long)LPAD * NKEY);
    softmax_rows<<<dim3(BATCH * 8, LPAD), 256, 0, stream>>>(scor, pmat);
    // O[bh] = P(48x1024) x V(1024x32)
    gemm_ab<false, true><<<dim3(2, 3, BATCH * 8), 32, 0, stream>>>(pmat, vf16, ofr,
        32, NKEY, (long long)LPAD * NKEY, (long long)NKEY * 32, (long long)LPAD * 32);

    merge_heads<<<dim3(BATCH, LPAD), 256, 0, stream>>>(ofr, om);
    gemm_aw<false><<<dim3(16, 3, BATCH), 32, 0, stream>>>(om, w_proj, xl,
        COUTC, COUTC, (long long)LPAD * COUTC, (long long)LPAD * COUTC);

    final_ln<<<dim3(BATCH, 301), 256, 0, stream>>>(xl, xg, g_ln, out);
}
